// LinearAttentionTransformer_14224931684795
// MI455X (gfx1250) — matrix operand order repaired and checked by reading, not compiled, not run on hardware
//
#include <hip/hip_runtime.h>
#include <cstdint>
#include <cstddef>

// ---------------- problem constants (from reference) ----------------
#define DEPTH   4
#define DIM     512
#define HEADS   8
#define SEQ     4096
#define BATCH   2
#define DH      64          // DIM / HEADS
#define BUCKET  64
#define FFDIM   (4 * DIM)   // 2048
#define MROWS   (BATCH * SEQ)  // 8192

typedef __bf16 bf16;
typedef bf16  v16bf __attribute__((ext_vector_type(16)));
typedef float v8f   __attribute__((ext_vector_type(8)));

// ---------------- helpers ----------------
__device__ __forceinline__ bf16 f2bf(float f) {
    union { float f; unsigned u; } a; a.f = f;
    unsigned r = a.u + 0x7FFFu + ((a.u >> 16) & 1u);   // round-to-nearest-even
    union { unsigned short s; bf16 b; } o; o.s = (unsigned short)(r >> 16);
    return o.b;
}

__device__ __forceinline__ v16bf ld_frag(const bf16* p0, const bf16* p1) {
    // 8 contiguous bf16 from p0 -> elements 0..7, 8 from p1 -> elements 8..15
    uint4 lo = *reinterpret_cast<const uint4*>(p0);
    uint4 hi = *reinterpret_cast<const uint4*>(p1);
    v16bf r;
    __builtin_memcpy(&r, &lo, 16);
    __builtin_memcpy(reinterpret_cast<char*>(&r) + 16, &hi, 16);
    return r;
}

__device__ __forceinline__ void store_val(float* p, float v) { *p = v; }
__device__ __forceinline__ void store_val(bf16*  p, float v) { *p = f2bf(v); }

// ---------------- weight prep: f32 [K,N] -> bf16 [N,K] (transposed) ----------------
__global__ __launch_bounds__(256) void wt_to_bf16t(const float* __restrict__ in,
                                                   bf16* __restrict__ out,
                                                   int K, int N) {
    int idx = blockIdx.x * 256 + threadIdx.x;
    if (idx >= K * N) return;
    int k = idx / N, n = idx - k * N;
    out[(size_t)n * K + k] = f2bf(in[idx]);
}

// ---------------- layernorm: f32 [M,512] -> bf16 [M,512] ----------------
__global__ __launch_bounds__(256) void layernorm_k(const float* __restrict__ x,
                                                   const float* __restrict__ g,
                                                   const float* __restrict__ b,
                                                   bf16* __restrict__ out) {
    __shared__ float red[256];
    int row = blockIdx.x, t = threadIdx.x;
    const float* xr = x + (size_t)row * DIM;
    float v0 = xr[t], v1 = xr[t + 256];
    red[t] = v0 + v1;
    __syncthreads();
    for (int o = 128; o > 0; o >>= 1) { if (t < o) red[t] += red[t + o]; __syncthreads(); }
    float mu = red[0] * (1.0f / DIM);
    __syncthreads();
    float d0 = v0 - mu, d1 = v1 - mu;
    red[t] = d0 * d0 + d1 * d1;
    __syncthreads();
    for (int o = 128; o > 0; o >>= 1) { if (t < o) red[t] += red[t + o]; __syncthreads(); }
    float rstd = rsqrtf(red[0] * (1.0f / DIM) + 1e-5f);
    bf16* orow = out + (size_t)row * DIM;
    orow[t]       = f2bf(d0 * rstd * g[t]       + b[t]);
    orow[t + 256] = f2bf(d1 * rstd * g[t + 256] + b[t + 256]);
}

// ---------------- bf16 WMMA GEMM:  C[M,N] = A[M,K](bf16) * Bt[N,K](bf16)^T ----------------
// Block: 256 threads = 8 waves, block tile 128(M) x 128(N).
// Waves: 4 along M x 2 along N; per-wave tile 32(M) x 64(N) = 2x4 WMMA 16x16 tiles.
// EPI: 0 = plain f32 out; 1 = +bias +residual, f32 out; 2 = +bias, exact GELU, bf16 out
template <int EPI, typename OutT>
__global__ __launch_bounds__(256) void gemm_bf16_wmma(
    const bf16* __restrict__ A, const bf16* __restrict__ Bt,
    const float* __restrict__ bias, const float* __restrict__ resid,
    OutT* __restrict__ out, int M, int N, int K)
{
    const int lane = threadIdx.x & 31;
    const int wave = threadIdx.x >> 5;
    const int m0 = blockIdx.x * 128 + (wave & 3) * 32;   // 4 waves along M
    const int n0 = blockIdx.y * 128 + (wave >> 2) * 64;  // 2 waves along N

    // A fragment addressing (16-bit A 16x32 layout):
    // lanes 0-15: row m0+lane, K {k0..k0+7, k0+16..k0+23}; lanes 16-31: K {+8..+15, +24..+31}
    const int half   = lane >> 4;
    const int ar     = m0 + (lane & 15);
    const int ak_off = half * 8;
    // B fragment addressing (32x16 layout, Bt is [N,K] row-major):
    // lanes 0-15: col n+lane, K {k0..k0+7, k0+16..k0+23}; lanes 16-31: col n+lane-16, K {+8..+15, +24..+31}
    const int bc     = n0 + (lane & 15);
    const int bk_off = half * 8;

    const bf16* ap[2];
    ap[0] = A + (size_t)ar * K + ak_off;
    ap[1] = A + (size_t)(ar + 16) * K + ak_off;
    const bf16* bp[4];
#pragma unroll
    for (int ni = 0; ni < 4; ++ni)
        bp[ni] = Bt + (size_t)(bc + ni * 16) * K + bk_off;

    v8f acc[2][4] = {};
    for (int k0 = 0; k0 < K; k0 += 32) {
        v16bf af[2], bfr[4];
#pragma unroll
        for (int mi = 0; mi < 2; ++mi) af[mi]  = ld_frag(ap[mi] + k0, ap[mi] + k0 + 16);
#pragma unroll
        for (int ni = 0; ni < 4; ++ni) bfr[ni] = ld_frag(bp[ni] + k0, bp[ni] + k0 + 16);
        if (k0 + 32 < K) {
            __builtin_prefetch(ap[0] + k0 + 32, 0, 3);
            __builtin_prefetch(bp[0] + k0 + 32, 0, 3);
        }
#pragma unroll
        for (int mi = 0; mi < 2; ++mi)
#pragma unroll
            for (int ni = 0; ni < 4; ++ni)
                acc[mi][ni] = __builtin_amdgcn_wmma_f32_16x16x32_bf16(
                    false, af[mi], false, bfr[ni], (short)0, acc[mi][ni], false, false);
    }

    // C/D layout: VGPR j: lanes 0-15 -> M=j, N=lane; lanes 16-31 -> M=8+j, N=lane-16
    const int rbase = (half << 3);
    const int cbase = (lane & 15);
#pragma unroll
    for (int mi = 0; mi < 2; ++mi)
#pragma unroll
        for (int ni = 0; ni < 4; ++ni)
#pragma unroll
            for (int j = 0; j < 8; ++j) {
                int row = m0 + mi * 16 + rbase + j;
                int col = n0 + ni * 16 + cbase;
                float v = acc[mi][ni][j];
                if (EPI >= 1) v += bias[col];
                if (EPI == 1) v += resid[(size_t)row * N + col];
                if (EPI == 2) v = 0.5f * v * (1.0f + erff(v * 0.70710678118f));
                store_val(out + (size_t)row * N + col, v);
            }
}

// ---------------- bucketed causal linear attention ----------------
// q,k,v: f32 [MROWS, DIM] with col = head*DH + e ; one block per (batch, head).
// Running state S = cumulative k'^T v (shifted one bucket), z = cumulative sum of k'.
__global__ __launch_bounds__(256) void attn_k(const float* __restrict__ q,
                                              const float* __restrict__ k,
                                              const float* __restrict__ v,
                                              bf16* __restrict__ att) {
    __shared__ float S[DH][DH + 1];        // running context
    __shared__ float z[DH];                // running key cumsum
    __shared__ float qk[BUCKET][DH + 1];   // phase A: q' ; update phase: exp(k)
    __shared__ float vv[BUCKET][DH + 1];
    __shared__ float Dinv[BUCKET];

    const int bh = blockIdx.x;
    const int batch = bh / HEADS, head = bh % HEADS;
    const int cb = head * DH;
    const size_t rowbase = (size_t)batch * SEQ;
    const int tid = threadIdx.x;
    const int ri = tid >> 2;               // row 0..63
    const int e0 = (tid & 3) << 4;         // 16-wide column slice

    for (int i = tid; i < DH * (DH + 1); i += 256) (&S[0][0])[i] = 0.0f;
    if (tid < DH) z[tid] = 0.0f;
    __syncthreads();

    for (int u = 0; u < SEQ / BUCKET; ++u) {
        const size_t row0 = rowbase + (size_t)u * BUCKET;
        // --- phase A1 (64 threads): softmax(q)/8 into qk, D and 1/clip(D) per row
        if (tid < BUCKET) {
            const float* qr = q + (row0 + tid) * DIM + cb;
            float mx = -3.4e38f;
            for (int j = 0; j < DH; ++j) { float t = qr[j]; qk[tid][j] = t; mx = fmaxf(mx, t); }
            float ssum = 0.0f;
            for (int j = 0; j < DH; ++j) { float t = __expf(qk[tid][j] - mx); qk[tid][j] = t; ssum += t; }
            const float scale = 0.125f / ssum;   // softmax * DH^-0.5 (64^-0.5 = 1/8)
            float D = 0.0f;
            for (int j = 0; j < DH; ++j) { float t = qk[tid][j] * scale; qk[tid][j] = t; D += t * z[j]; }
            Dinv[tid] = 1.0f / fmaxf(D, 1e-3f);
        }
        __syncthreads();
        // --- phase A2 (all 256 threads): out[ri][e0..e0+15] = (q' @ S) * Dinv
        {
            float o[16];
#pragma unroll
            for (int e = 0; e < 16; ++e) o[e] = 0.0f;
            for (int d = 0; d < DH; ++d) {
                const float qd = qk[ri][d];
#pragma unroll
                for (int e = 0; e < 16; ++e) o[e] += qd * S[d][e0 + e];
            }
            bf16* orow = att + (row0 + ri) * DIM + cb + e0;
            const float di = Dinv[ri];
#pragma unroll
            for (int e = 0; e < 16; ++e) orow[e] = f2bf(o[e] * di);
        }
        __syncthreads();
        // --- stage exp(k) (reusing qk) and v for this bucket
        for (int t = tid; t < BUCKET * DH; t += 256) {
            int i = t >> 6, j = t & 63;
            qk[i][j] = __expf(k[(row0 + i) * DIM + cb + j]);
            vv[i][j] = v[(row0 + i) * DIM + cb + j];
        }
        __syncthreads();
        // --- update S (outer-product accumulate over the bucket) and z
        {
            float accs[16];
#pragma unroll
            for (int e = 0; e < 16; ++e) accs[e] = 0.0f;
            float zacc = 0.0f;
            for (int i = 0; i < BUCKET; ++i) {
                const float kv = qk[i][ri];
                zacc += kv;
#pragma unroll
                for (int e = 0; e < 16; ++e) accs[e] += kv * vv[i][e0 + e];
            }
#pragma unroll
            for (int e = 0; e < 16; ++e) S[ri][e0 + e] += accs[e];
            if ((tid & 3) == 0) z[ri] += zacc;
        }
        __syncthreads();
    }
}

// ---------------- host orchestration ----------------
extern "C" void kernel_launch(void* const* d_in, const int* in_sizes, int n_in,
                              void* d_out, int out_size, void* d_ws, size_t ws_size,
                              hipStream_t stream)
{
    const float* x_in  = (const float*)d_in[0];
    const float* ln1_g = (const float*)d_in[1];
    const float* ln1_b = (const float*)d_in[2];
    const float* wq    = (const float*)d_in[3];
    const float* wk    = (const float*)d_in[4];
    const float* wv    = (const float*)d_in[5];
    const float* wo    = (const float*)d_in[6];
    const float* bo    = (const float*)d_in[7];
    const float* ln2_g = (const float*)d_in[8];
    const float* ln2_b = (const float*)d_in[9];
    const float* w1    = (const float*)d_in[10];
    const float* b1    = (const float*)d_in[11];
    const float* w2    = (const float*)d_in[12];
    const float* b2    = (const float*)d_in[13];

    const size_t NP = (size_t)DIM * DIM;     // proj weight elems
    const size_t NF = (size_t)DIM * FFDIM;   // ff weight elems

    char* p = (char*)d_ws;
    size_t used = 0;
    auto carve = [&](size_t bytes) {
        char* r = p;
        size_t pad = (bytes + 255) & ~(size_t)255;
        p += pad; used += pad;
        return (void*)r;
    };
    bf16*  wqt   = (bf16*)carve(DEPTH * NP * sizeof(bf16));
    bf16*  wkt   = (bf16*)carve(DEPTH * NP * sizeof(bf16));
    bf16*  wvt   = (bf16*)carve(DEPTH * NP * sizeof(bf16));
    bf16*  wot   = (bf16*)carve(DEPTH * NP * sizeof(bf16));
    bf16*  w1t   = (bf16*)carve(DEPTH * NF * sizeof(bf16));
    bf16*  w2t   = (bf16*)carve(DEPTH * NF * sizeof(bf16));
    float* x_ws  = (float*)carve((size_t)MROWS * DIM * sizeof(float));
    bf16*  h_bf  = (bf16*)carve((size_t)MROWS * DIM * sizeof(bf16));
    float* qf    = (float*)carve((size_t)MROWS * DIM * sizeof(float));
    float* kf    = (float*)carve((size_t)MROWS * DIM * sizeof(float));
    float* vf    = (float*)carve((size_t)MROWS * DIM * sizeof(float));
    bf16*  attbf = (bf16*)carve((size_t)MROWS * DIM * sizeof(bf16));
    bf16*  ffbf  = (bf16*)carve((size_t)MROWS * FFDIM * sizeof(bf16));
    if (used > ws_size) return;   // workspace too small: bail deterministically

    // ---- weight prep (bf16, transposed to [N,K]) ----
    const int bp = (int)((NP + 255) / 256);
    const int bf = (int)((NF + 255) / 256);
    for (int i = 0; i < DEPTH; ++i) {
        wt_to_bf16t<<<bp, 256, 0, stream>>>(wq + i * NP, wqt + i * NP, DIM, DIM);
        wt_to_bf16t<<<bp, 256, 0, stream>>>(wk + i * NP, wkt + i * NP, DIM, DIM);
        wt_to_bf16t<<<bp, 256, 0, stream>>>(wv + i * NP, wvt + i * NP, DIM, DIM);
        wt_to_bf16t<<<bp, 256, 0, stream>>>(wo + i * NP, wot + i * NP, DIM, DIM);
        wt_to_bf16t<<<bf, 256, 0, stream>>>(w1 + i * NF, w1t + i * NF, DIM, FFDIM);
        wt_to_bf16t<<<bf, 256, 0, stream>>>(w2 + i * NF, w2t + i * NF, FFDIM, DIM);
    }

    const dim3 gD(MROWS / 128, DIM / 128);    // GEMMs with N = 512
    const dim3 gF(MROWS / 128, FFDIM / 128);  // FF1 with N = 2048

    const float* xcur = x_in;
    for (int i = 0; i < DEPTH; ++i) {
        // PreNorm attention
        layernorm_k<<<MROWS, 256, 0, stream>>>(xcur, ln1_g + i * DIM, ln1_b + i * DIM, h_bf);
        gemm_bf16_wmma<0, float><<<gD, 256, 0, stream>>>(h_bf, wqt + i * NP, nullptr, nullptr, qf, MROWS, DIM, DIM);
        gemm_bf16_wmma<0, float><<<gD, 256, 0, stream>>>(h_bf, wkt + i * NP, nullptr, nullptr, kf, MROWS, DIM, DIM);
        gemm_bf16_wmma<0, float><<<gD, 256, 0, stream>>>(h_bf, wvt + i * NP, nullptr, nullptr, vf, MROWS, DIM, DIM);
        attn_k<<<BATCH * HEADS, 256, 0, stream>>>(qf, kf, vf, attbf);
        gemm_bf16_wmma<1, float><<<gD, 256, 0, stream>>>(attbf, wot + i * NP, bo + i * DIM, xcur, x_ws, MROWS, DIM, DIM);
        // PreNorm feed-forward
        layernorm_k<<<MROWS, 256, 0, stream>>>(x_ws, ln2_g + i * DIM, ln2_b + i * DIM, h_bf);
        gemm_bf16_wmma<2, bf16><<<gF, 256, 0, stream>>>(h_bf, w1t + i * NF, b1 + i * FFDIM, nullptr, ffbf, MROWS, FFDIM, DIM);
        float* xo = (i == DEPTH - 1) ? (float*)d_out : x_ws;
        gemm_bf16_wmma<1, float><<<gD, 256, 0, stream>>>(ffbf, w2t + i * NF, b2 + i * DIM, x_ws, xo, MROWS, DIM, FFDIM);
        xcur = x_ws;
    }
}
